// LSTM_model_59244778881379
// MI455X (gfx1250) — matrix operand order repaired and checked by reading, not compiled, not run on hardware
//
#include <hip/hip_runtime.h>
#include <hip/hip_bf16.h>

// ---------------------------------------------------------------------------
// Autoregressive LSTM rollout on gfx1250 using v_wmma_f32_16x16x32_bf16.
//   B=8192, HORIZON=64, F=64, ORDER=16, K=256 (4K=1024 gate columns)
// One workgroup = 64 batch rows (4 waves), runs all 64 timesteps.
// Partition: each wave owns 4 unit tiles (64 units x 4 gates) across ALL
// 64 batch rows -> B fragments reused 4x (M-tiles), A fragments reused 4x
// (gates); weights stream from L2 exactly once per WG per timestep.
// ---------------------------------------------------------------------------

typedef __bf16 v16bf __attribute__((ext_vector_type(16)));
typedef float  v8f   __attribute__((ext_vector_type(8)));

#define B_TOT    8192
#define HORIZON  64
#define F_IN     64
#define ORDER    16
#define KU       256     // hidden units
#define GATES4   1024    // 4*KU
#define INP_COLS 96      // F(64) + ORDER(16) + pad(16) -> 3 K-chunks of 32
#define KROWS    352     // 96 (padded input) + 256 (h)
#define KCHUNKS  11      // 352 / 32
#define BM       64      // batch rows per workgroup (4 M-tiles of 16)
#define THREADS  128     // 4 waves (wave32)
#define UT_PER_WAVE 4    // unit tiles per wave (4 waves * 4 = 16 = KU/16)

static __device__ __forceinline__ unsigned short f32_bf16(float f) {
  union { float f; unsigned u; } x; x.f = f;
  unsigned r = x.u + 0x7FFFu + ((x.u >> 16) & 1u);   // round-to-nearest-even
  return (unsigned short)(r >> 16);
}
static __device__ __forceinline__ float sigm(float x) {
  return 1.0f / (1.0f + __expf(-x));
}
static __device__ __forceinline__ float tanh_fast(float x) {
  float e = __expf(2.0f * x);
  return (e - 1.0f) / (e + 1.0f);
}

// ---------------------------------------------------------------------------
// Prep: pack [kernel(80x1024); zeros(16x1024); rec_kernel(256x1024)] (f32)
// into bf16 fragments. Fragment (kc, ntile) = 32 lanes x 16 bf16 = 1KB,
// element (lane, j) = W[kc*32 + 8*(lane/16) + (j&7) + 16*(j>>3)][ntile*16 + lane%16]  (B-matrix lane layout).
// ---------------------------------------------------------------------------
__global__ void pack_weights(const float* __restrict__ kern,
                             const float* __restrict__ rec,
                             unsigned short* __restrict__ wp) {
  int idx = blockIdx.x * blockDim.x + threadIdx.x;
  if (idx >= KROWS * GATES4) return;
  int k   = idx >> 10;       // 0..351
  int col = idx & 1023;      // 0..1023
  float v;
  if (k < F_IN + ORDER)      v = kern[k * GATES4 + col];
  else if (k < INP_COLS)     v = 0.0f;                       // pad rows
  else                       v = rec[(k - INP_COLS) * GATES4 + col];
  int kc = k >> 5, kl = k & 31;
  int ntile = col >> 4;
  int lane = (col & 15) | (((kl >> 3) & 1) << 4), j = (kl & 7) | ((kl >> 4) << 3);
  wp[(((kc << 6) | ntile) << 9) + (lane << 4) + j] = f32_bf16(v);
}

// ---------------------------------------------------------------------------
// Main rollout kernel.
// ---------------------------------------------------------------------------
__global__ void __launch_bounds__(THREADS)
lstm_rollout(const float* __restrict__ x,        // [B,64,64]
             const float* __restrict__ y0,       // [B,16]
             const unsigned short* __restrict__ wp,
             const float* __restrict__ bias,     // [1024]
             const float* __restrict__ dw,       // [256]
             const float* __restrict__ db,       // [1]
             float* __restrict__ out) {          // [B,64]
  extern __shared__ char smem_raw[];
  unsigned short* inp = (unsigned short*)smem_raw;          // [64][96]  bf16
  unsigned short* hA  = inp + BM * INP_COLS;                // [64][256] bf16
  unsigned short* hB  = hA + BM * KU;                       // [64][256] bf16
  float*          cst = (float*)(hB + BM * KU);             // [64][256] f32
  float*          yp  = cst + BM * KU;                      // [64][16]  f32
  float*          dwL = yp + BM * ORDER;                    // [256]     f32
  float*          part = dwL + KU;                          // [128]     f32

  const int tid  = threadIdx.x;
  const int wave = tid >> 5;
  const int lane = tid & 31;
  const int b0   = blockIdx.x * BM;

  // ---- init: h=0, c=0, yp=y0, dense_w staged, pad columns of inp = 0 ----
  for (int i = tid; i < BM * KU; i += THREADS) { hA[i] = 0; cst[i] = 0.0f; }
  for (int i = tid; i < BM * ORDER; i += THREADS)
    yp[i] = y0[(size_t)(b0 + (i >> 4)) * ORDER + (i & 15)];
  for (int i = tid; i < KU; i += THREADS) dwL[i] = dw[i];
  for (int r = tid; r < BM; r += THREADS)
    for (int j = F_IN + ORDER; j < INP_COLS; ++j) inp[r * INP_COLS + j] = 0;
  __syncthreads();

  const int   halfsel = lane >> 4;       // 0: lanes 0-15, 1: lanes 16-31
  const int   colN    = lane & 15;       // C/D column within tile; A row M
  const float dbv     = db[0];

  for (int t = 0; t < HORIZON; ++t) {
    // ---- stage x_t (f32 -> bf16) and yp into inp ----
    for (int i = tid; i < BM * F_IN; i += THREADS) {
      int r = i >> 6, c = i & 63;
      inp[r * INP_COLS + c] =
          f32_bf16(x[(size_t)(b0 + r) * (HORIZON * F_IN) + t * F_IN + c]);
    }
    for (int i = tid; i < BM * ORDER; i += THREADS) {
      int r = i >> 4, j = i & 15;
      inp[r * INP_COLS + F_IN + j] = f32_bf16(yp[r * ORDER + j]);
    }
    __syncthreads();

    const unsigned short* hRead  = (t & 1) ? hB : hA;
    unsigned short*       hWrite = (t & 1) ? hA : hB;

    // ---- per-wave: 4 unit tiles x (4 M-tiles x 4 gates) accumulators ----
    for (int ut = 0; ut < UT_PER_WAVE; ++ut) {
      const int gut = wave * UT_PER_WAVE + ut;   // global unit tile 0..15

      v8f acc[4][4];                             // [mtile][gate]
#pragma unroll
      for (int g = 0; g < 4; ++g) {
        float bv = bias[g * KU + gut * 16 + colN];
#pragma unroll
        for (int m = 0; m < 4; ++m)
          acc[m][g] = (v8f){bv, bv, bv, bv, bv, bv, bv, bv};
      }

#pragma unroll
      for (int kc = 0; kc < KCHUNKS; ++kc) {
        // ---- 4 A fragments (one per M-tile) from LDS activations ----
        union { v16bf v; uint4 q[2]; } af[4];
#pragma unroll
        for (int m = 0; m < 4; ++m) {
          const int row_local = m * 16 + colN;   // A-matrix row (M = lane%16)
          const unsigned short* rowp;
          int kb;
          if (kc < 3) { rowp = inp + row_local * INP_COLS; kb = kc * 32; }
          else        { rowp = hRead + row_local * KU;     kb = (kc - 3) * 32; }
          af[m].q[0] = *(const uint4*)(rowp + kb + halfsel * 8);
          af[m].q[1] = *(const uint4*)(rowp + kb + 16 + halfsel * 8);
        }
        // ---- 4 B fragments (one per gate), hoisted ahead of WMMA burst ----
        union { v16bf v; uint4 q[2]; } bfr[4];
#pragma unroll
        for (int g = 0; g < 4; ++g) {
          int ntile = g * 16 + gut;              // column tile 0..63
          const uint4* bp =
              (const uint4*)(wp + (((kc << 6) | ntile) << 9) + (lane << 4));
          bfr[g].q[0] = bp[0];
          bfr[g].q[1] = bp[1];
        }
        // ---- 16 independent WMMAs ----
#pragma unroll
        for (int m = 0; m < 4; ++m)
#pragma unroll
          for (int g = 0; g < 4; ++g)
            acc[m][g] = __builtin_amdgcn_wmma_f32_16x16x32_bf16(
                false, af[m].v, false, bfr[g].v, (short)0, acc[m][g],
                false, false);
      }

      // ---- gate fusion in registers (lane-identical C layouts) ----
#pragma unroll
      for (int m = 0; m < 4; ++m) {
#pragma unroll
        for (int v = 0; v < 8; ++v) {
          int row = m * 16 + v + halfsel * 8;    // batch row 0..63
          int idx = row * KU + gut * 16 + colN;
          float iv = sigm(acc[m][0][v]);
          float fv = sigm(acc[m][1][v]);
          float gv = tanh_fast(acc[m][2][v]);
          float ov = sigm(acc[m][3][v]);
          float cn = fv * cst[idx] + iv * gv;
          cst[idx] = cn;
          hWrite[idx] = f32_bf16(ov * tanh_fast(cn));
        }
      }
    }
    __syncthreads();

    // ---- dense head: all 128 threads, 2 per row, vectorized bf16 loads ----
    {
      const int r = tid & 63, half = tid >> 6;
      const unsigned short* hr = hWrite + r * KU + half * 128;
      const float* dwp = dwL + half * 128;
      float s = 0.0f;
#pragma unroll
      for (int k0 = 0; k0 < 128; k0 += 8) {
        uint4 q = *(const uint4*)(hr + k0);
        const unsigned* qu = (const unsigned*)&q;
#pragma unroll
        for (int e = 0; e < 4; ++e) {
          union { unsigned u; float f; } lo, hi;
          lo.u = qu[e] << 16;                 // bf16 elem 2e   -> f32
          hi.u = qu[e] & 0xFFFF0000u;         // bf16 elem 2e+1 -> f32 (free)
          s += lo.f * dwp[k0 + 2 * e] + hi.f * dwp[k0 + 2 * e + 1];
        }
      }
      part[tid] = s;
    }
    __syncthreads();

    // ---- finalize pred + autoregressive feedback shift ----
    if (tid < BM) {
      float s = dbv + part[tid] + part[tid + 64];
      out[(size_t)(b0 + tid) * HORIZON + t] = s;
      float tmp[ORDER - 1];
#pragma unroll
      for (int j = 0; j < ORDER - 1; ++j) tmp[j] = yp[tid * ORDER + j];
      yp[tid * ORDER] = s;
#pragma unroll
      for (int j = 0; j < ORDER - 1; ++j) yp[tid * ORDER + 1 + j] = tmp[j];
    }
    __syncthreads();
  }
}

extern "C" void kernel_launch(void* const* d_in, const int* in_sizes, int n_in,
                              void* d_out, int out_size, void* d_ws, size_t ws_size,
                              hipStream_t stream) {
  (void)in_sizes; (void)n_in; (void)out_size; (void)ws_size;
  const float* x    = (const float*)d_in[0];
  const float* y0   = (const float*)d_in[1];
  const float* kern = (const float*)d_in[2];
  const float* rec  = (const float*)d_in[3];
  const float* bias = (const float*)d_in[4];
  const float* dw   = (const float*)d_in[5];
  const float* db   = (const float*)d_in[6];
  unsigned short* wp = (unsigned short*)d_ws;   // 352*1024*2 = 704 KB

  // 1) pack weights to bf16 fragment-major layout (L2-resident afterwards)
  int total = KROWS * GATES4;
  pack_weights<<<(total + 255) / 256, 256, 0, stream>>>(kern, rec, wp);

  // 2) rollout: 128 workgroups x 128 threads, ~145 KB dynamic LDS each
  //    (gfx1250 WGP has 320 KB LDS; opt in above the 64 KB default)
  size_t lds = (size_t)(BM * INP_COLS + 2 * BM * KU) * sizeof(unsigned short)
             + (size_t)(BM * KU + BM * ORDER + KU + THREADS) * sizeof(float);
  hipFuncSetAttribute(reinterpret_cast<const void*>(lstm_rollout),
                      hipFuncAttributeMaxDynamicSharedMemorySize, (int)lds);
  lstm_rollout<<<B_TOT / BM, THREADS, lds, stream>>>(
      x, y0, wp, bias, dw, db, (float*)d_out);
}
